// GRU_sememe_54571854463109
// MI455X (gfx1250) — matrix operand order repaired and checked by reading, not compiled, not run on hardware
//
#include <hip/hip_runtime.h>

// ---------------------------------------------------------------------------
// Problem constants (from reference): T=256, B=64, NINP=NHID=512
// ---------------------------------------------------------------------------
constexpr int T_STEPS = 256;
constexpr int BATCH   = 64;
constexpr int NINP    = 512;
constexpr int NHID    = 512;
constexpr int MROWS   = T_STEPS * BATCH;   // 16384
constexpr int KDIM    = 2 * NINP;          // 1024 (concat input dim)
constexpr int NIOU    = 2 * NHID;          // 1024

typedef __bf16 bf16_t;
typedef __attribute__((ext_vector_type(16))) __bf16 v16bf;
typedef __attribute__((ext_vector_type(8)))  __bf16 v8bf;
typedef __attribute__((ext_vector_type(8)))  float  v8f;

__device__ __forceinline__ v16bf cat16(v8bf lo, v8bf hi) {
  return __builtin_shufflevector(lo, hi, 0, 1, 2, 3, 4, 5, 6, 7,
                                 8, 9, 10, 11, 12, 13, 14, 15);
}

__device__ __forceinline__ v8f vzero8() {
  v8f z;
#pragma unroll
  for (int i = 0; i < 8; ++i) z[i] = 0.0f;
  return z;
}

__device__ __forceinline__ float sigmoidf_(float x) {
  return 1.0f / (1.0f + __expf(-x));
}

// ---------------------------------------------------------------------------
// gfx1250 async global->LDS staging (ASYNCcnt path). The builtin's param 0 is
// 'v4i __device__*' (int vector_size(16), AS(1)) per compiler diagnostics, so
// cast through exactly that type; LDS side through the AS(3) counterpart.
// ---------------------------------------------------------------------------
#if defined(__has_builtin)
#if __has_builtin(__builtin_amdgcn_global_load_async_to_lds_b128)
#define HAS_ASYNC_LDS 1
#endif
#endif

#ifdef HAS_ASYNC_LDS
typedef int v4i __attribute__((vector_size(16)));
typedef __attribute__((address_space(1))) v4i* gp_v4i;
typedef __attribute__((address_space(3))) v4i* lp_v4i;

__device__ __forceinline__ void async_copy16(const void* g, void* l) {
  __builtin_amdgcn_global_load_async_to_lds_b128((gp_v4i)g, (lp_v4i)l, 0, 0);
}
__device__ __forceinline__ void async_wait0() {
#if __has_builtin(__builtin_amdgcn_s_wait_asynccnt)
  __builtin_amdgcn_s_wait_asynccnt(0);
#else
  asm volatile("s_wait_asynccnt 0x0" ::: "memory");
#endif
}
#endif

// ---------------------------------------------------------------------------
// x = concat(inputs, sememe_h) -> bf16, [M=T*B, K=1024], float4-vectorized.
// A 4-wide chunk never straddles the 512 concat boundary.
// ---------------------------------------------------------------------------
__global__ void k_build_x(const float* __restrict__ inputs,
                          const float* __restrict__ sememe,
                          bf16_t* __restrict__ x) {
  int idx = (blockIdx.x * blockDim.x + threadIdx.x) * 4;   // over MROWS*KDIM
  int row = idx >> 10;
  int col = idx & (KDIM - 1);
  const float* src = (col < NINP) ? (inputs + row * NINP + col)
                                  : (sememe + row * NINP + (col - NINP));
  float4 v = *(const float4*)src;
  x[idx + 0] = (bf16_t)v.x;
  x[idx + 1] = (bf16_t)v.y;
  x[idx + 2] = (bf16_t)v.z;
  x[idx + 3] = (bf16_t)v.w;
}

__global__ void k_to_bf16(const float* __restrict__ s,
                          bf16_t* __restrict__ d, int n) {
  int i = (blockIdx.x * blockDim.x + threadIdx.x) * 4;
  if (i < n) {
    float4 v = *(const float4*)(s + i);
    d[i + 0] = (bf16_t)v.x;
    d[i + 1] = (bf16_t)v.y;
    d[i + 2] = (bf16_t)v.z;
    d[i + 3] = (bf16_t)v.w;
  }
}

__global__ void k_init_h(const float* __restrict__ hx,
                         float* __restrict__ h_cur,
                         bf16_t* __restrict__ h_bf) {
  int i = blockIdx.x * blockDim.x + threadIdx.x;     // BATCH*NHID
  float v = hx[i];
  h_cur[i] = v;
  h_bf[i]  = (bf16_t)v;
}

// ---------------------------------------------------------------------------
// Big hoisted GEMM: C[M,N] = X[M,K] * W[N,K]^T + bias[N]
//   M = 16384, K = 1024, N in {1024, 512}. bf16 in, f32 accum/out.
// 128x128 macrotile / 8-wave workgroup; K staged 32/iter into DOUBLE-BUFFERED
// LDS so next-tile traffic (async-to-LDS when available) overlaps the WMMAs.
// Wave w owns rows [16w,16w+15] x 128 cols = 8 accumulators; A-fragment is
// loaded once per K-chunk; B-fragments batched 4 at a time -> 4 back-to-back
// v_wmma issues per group.
// ---------------------------------------------------------------------------
__global__ __launch_bounds__(256) void k_gemm_xw(
    const bf16_t* __restrict__ X,     // [M, K] row-major
    const bf16_t* __restrict__ W,     // [N, K] row-major
    const float*  __restrict__ bias,  // [N]
    float* __restrict__ C,            // [M, N]
    int N) {
  constexpr int K   = KDIM;
  constexpr int NK  = K / 32;
  constexpr int LDA = 40;                 // padded bf16 elems per LDS row
  __shared__ bf16_t Alds[2][128 * LDA];   // 2 x 10 KB
  __shared__ bf16_t Blds[2][128 * LDA];   // 2 x 10 KB

  const int tid  = threadIdx.x;
  const int wave = tid >> 5;              // 0..7 (wave32)
  const int lane = tid & 31;
  const int hlf  = lane >> 4;
  const int l16  = lane & 15;
  const int tile_m = blockIdx.y * 128;
  const int tile_n = blockIdx.x * 128;

  v8f acc[8];
#pragma unroll
  for (int i = 0; i < 8; ++i) acc[i] = vzero8();

  // Cooperative fill mapping: each thread owns 32B of A and 32B of B per step
  const int fill_row = tid >> 1;          // 0..127
  const int fill_chk = (tid & 1) * 16;    // element offset 0 or 16
  const bf16_t* gA = X + (size_t)(tile_m + fill_row) * K + fill_chk;
  const bf16_t* gB = W + (size_t)(tile_n + fill_row) * K + fill_chk;
  bf16_t* sA0 = &Alds[0][fill_row * LDA + fill_chk];
  bf16_t* sA1 = &Alds[1][fill_row * LDA + fill_chk];
  bf16_t* sB0 = &Blds[0][fill_row * LDA + fill_chk];
  bf16_t* sB1 = &Blds[1][fill_row * LDA + fill_chk];

#ifdef HAS_ASYNC_LDS
  auto stage = [&](int b, int kb) {
    bf16_t* la = b ? sA1 : sA0;
    bf16_t* lb = b ? sB1 : sB0;
    async_copy16(gA + kb,     la);
    async_copy16(gA + kb + 8, la + 8);
    async_copy16(gB + kb,     lb);
    async_copy16(gB + kb + 8, lb + 8);
  };
  stage(0, 0);
  async_wait0();
  __syncthreads();
#else
  uint4 ra0, ra1, rb0, rb1;
  auto fetch = [&](int kb) {
    const uint4* pa = (const uint4*)(gA + kb);
    const uint4* pb = (const uint4*)(gB + kb);
    ra0 = pa[0]; ra1 = pa[1];
    rb0 = pb[0]; rb1 = pb[1];
  };
  auto publish = [&](int b) {
    uint4* la = (uint4*)(b ? sA1 : sA0);
    uint4* lb = (uint4*)(b ? sB1 : sB0);
    la[0] = ra0; la[1] = ra1;
    lb[0] = rb0; lb[1] = rb1;
  };
  fetch(0);
  publish(0);
  __syncthreads();
#endif

  for (int it = 0; it < NK; ++it) {
    const int cur = it & 1;
#ifdef HAS_ASYNC_LDS
    if (it + 1 < NK) stage(cur ^ 1, (it + 1) * 32);   // in flight during WMMAs
#else
    if (it + 1 < NK) fetch((it + 1) * 32);            // loads in flight
#endif

    // ---- A fragment (16x32 bf16, contiguous K: k = 16*hlf + s) ----
    const bf16_t* arow = &Alds[cur][(wave * 16 + l16) * LDA];
    v16bf afrag = cat16(*(const v8bf*)(arow + 16 * hlf),
                        *(const v8bf*)(arow + 16 * hlf + 8));

    // ---- B fragments in two groups of 4 -> back-to-back WMMA issue ----
#pragma unroll
    for (int g = 0; g < 2; ++g) {
      v16bf bfrag[4];
#pragma unroll
      for (int j = 0; j < 4; ++j) {
        const bf16_t* brow =
            &Blds[cur][((g * 4 + j) * 16 + l16) * LDA + hlf * 16];
        bfrag[j] = cat16(*(const v8bf*)(brow), *(const v8bf*)(brow + 8));
      }
#pragma unroll
      for (int j = 0; j < 4; ++j) {
        acc[g * 4 + j] = __builtin_amdgcn_wmma_f32_16x16x32_bf16(
            false, afrag, false, bfrag[j], (short)0, acc[g * 4 + j], false,
            false);
      }
    }

#ifdef HAS_ASYNC_LDS
    if (it + 1 < NK) async_wait0();   // next tile landed in LDS
#else
    if (it + 1 < NK) publish(cur ^ 1);
#endif
    __syncthreads();
  }

  // ---- epilogue: C = acc + bias ----
#pragma unroll
  for (int nt = 0; nt < 8; ++nt) {
    int n = tile_n + nt * 16 + l16;
    float bv = bias[n];
#pragma unroll
    for (int r = 0; r < 8; ++r) {
      int m = tile_m + wave * 16 + r + 8 * hlf;
      C[(size_t)m * N + n] = acc[nt][r] + bv;
    }
  }
}

// ---------------------------------------------------------------------------
// Recurrent step, phase 1: iou = iou_x[t] + h @ W_iouh^T + b_iouh
//   z = sigmoid(iou[:, :512])  -> z_buf
//   r = sigmoid(iou[:, 512:]); rh = r * h -> rh_bf (bf16 A for phase 2)
// M=64, N=1024, K=512. Each wave owns 1x4 tiles (A reused 4x, 4 independent
// WMMA chains). Weights stay resident in the 192MB L2 across the scan.
// ---------------------------------------------------------------------------
__global__ __launch_bounds__(128) void k_step_iou(
    const bf16_t* __restrict__ Hbf,    // [64, 512] bf16
    const bf16_t* __restrict__ Wiouh,  // [1024, 512] bf16 row-major
    const float*  __restrict__ b_iouh, // [1024]
    const float*  __restrict__ ioux_t, // [64, 1024] f32 (this step's slice)
    const float*  __restrict__ h_cur,  // [64, 512] f32
    float* __restrict__ z_buf,         // [64, 512]
    bf16_t* __restrict__ rh_bf) {      // [64, 512]
  constexpr int K = NHID;
  constexpr int N = NIOU;
  const int wave = threadIdx.x >> 5;
  const int lane = threadIdx.x & 31;
  const int hlf  = lane >> 4;
  const int l16  = lane & 15;
  const int id      = blockIdx.x * 4 + wave;  // 0..63
  const int tile_m  = (id >> 4) * 16;         // 4 M tiles
  const int tile_n0 = (id & 15) * 64;         // 16 groups of 64 cols

  v8f acc[4];
#pragma unroll
  for (int i = 0; i < 4; ++i) acc[i] = vzero8();

  const bf16_t* arow  = Hbf + (size_t)(tile_m + l16) * K;
  const bf16_t* brow0 = Wiouh + (size_t)(tile_n0 + l16) * K + hlf * 16;

  for (int kb = 0; kb < K; kb += 32) {
    v16bf afrag = cat16(*(const v8bf*)(arow + kb + 16 * hlf),
                        *(const v8bf*)(arow + kb + 16 * hlf + 8));
#pragma unroll
    for (int nt = 0; nt < 4; ++nt) {
      const bf16_t* bb = brow0 + (size_t)nt * 16 * K + kb;
      v16bf bfrag = cat16(*(const v8bf*)bb, *(const v8bf*)(bb + 8));
      acc[nt] = __builtin_amdgcn_wmma_f32_16x16x32_bf16(
          false, afrag, false, bfrag, (short)0, acc[nt], false, false);
    }
  }

#pragma unroll
  for (int nt = 0; nt < 4; ++nt) {
#pragma unroll
    for (int r = 0; r < 8; ++r) {
      int m = tile_m + r + 8 * hlf;
      int n = tile_n0 + nt * 16 + l16;
      float v = acc[nt][r] + ioux_t[m * N + n] + b_iouh[n];
      float s = sigmoidf_(v);
      if (n < NHID) {                 // uniform: 64-col group stays one side
        z_buf[m * NHID + n] = s;
      } else {
        int nn = n - NHID;
        rh_bf[m * NHID + nn] = (bf16_t)(s * h_cur[m * NHID + nn]);
      }
    }
  }
}

// ---------------------------------------------------------------------------
// Recurrent step, phase 2: h_tilde = tanh(fx_x[t] + rh @ W_Uh^T + b_Uh)
//   h_new = (1-z)*h + z*h_tilde -> out[t], h_cur, h_bf
// M=64, N=512, K=512. 1x4 tiles per wave.
// ---------------------------------------------------------------------------
__global__ __launch_bounds__(128) void k_step_h(
    const bf16_t* __restrict__ RHbf,  // [64, 512] bf16
    const bf16_t* __restrict__ WUh,   // [512, 512] bf16 row-major
    const float*  __restrict__ b_Uh,  // [512]
    const float*  __restrict__ fxx_t, // [64, 512] f32 (this step's slice)
    const float*  __restrict__ z_buf, // [64, 512]
    float* __restrict__ h_cur,        // [64, 512] in/out
    bf16_t* __restrict__ h_bf,        // [64, 512] out (next step's A)
    float* __restrict__ out_t) {      // [64, 512] -> d_out slice
  constexpr int K = NHID;
  constexpr int N = NHID;
  const int wave = threadIdx.x >> 5;
  const int lane = threadIdx.x & 31;
  const int hlf  = lane >> 4;
  const int l16  = lane & 15;
  const int id      = blockIdx.x * 4 + wave;  // 0..31
  const int tile_m  = (id >> 3) * 16;         // 4 M tiles
  const int tile_n0 = (id & 7) * 64;          // 8 groups of 64 cols

  v8f acc[4];
#pragma unroll
  for (int i = 0; i < 4; ++i) acc[i] = vzero8();

  const bf16_t* arow  = RHbf + (size_t)(tile_m + l16) * K;
  const bf16_t* brow0 = WUh + (size_t)(tile_n0 + l16) * K + hlf * 16;

  for (int kb = 0; kb < K; kb += 32) {
    v16bf afrag = cat16(*(const v8bf*)(arow + kb + 16 * hlf),
                        *(const v8bf*)(arow + kb + 16 * hlf + 8));
#pragma unroll
    for (int nt = 0; nt < 4; ++nt) {
      const bf16_t* bb = brow0 + (size_t)nt * 16 * K + kb;
      v16bf bfrag = cat16(*(const v8bf*)bb, *(const v8bf*)(bb + 8));
      acc[nt] = __builtin_amdgcn_wmma_f32_16x16x32_bf16(
          false, afrag, false, bfrag, (short)0, acc[nt], false, false);
    }
  }

#pragma unroll
  for (int nt = 0; nt < 4; ++nt) {
#pragma unroll
    for (int r = 0; r < 8; ++r) {
      int m = tile_m + r + 8 * hlf;
      int n = tile_n0 + nt * 16 + l16;
      float u  = acc[nt][r] + fxx_t[m * N + n] + b_Uh[n];
      float ht = tanhf(u);
      float z  = z_buf[m * N + n];
      float h0 = h_cur[m * N + n];
      float hn = (1.0f - z) * h0 + z * ht;
      out_t[m * N + n] = hn;
      h_cur[m * N + n] = hn;
      h_bf[m * N + n]  = (bf16_t)hn;
    }
  }
}

// ---------------------------------------------------------------------------
// Host side
// ---------------------------------------------------------------------------
extern "C" void kernel_launch(void* const* d_in, const int* in_sizes, int n_in,
                              void* d_out, int out_size, void* d_ws,
                              size_t ws_size, hipStream_t stream) {
  (void)in_sizes; (void)n_in; (void)out_size; (void)ws_size;
  const float* inputs  = (const float*)d_in[0];
  const float* sememe  = (const float*)d_in[1];
  const float* hx      = (const float*)d_in[2];
  const float* W_ioux  = (const float*)d_in[3];
  const float* b_ioux  = (const float*)d_in[4];
  const float* W_iouh  = (const float*)d_in[5];
  const float* b_iouh  = (const float*)d_in[6];
  const float* W_fx    = (const float*)d_in[7];
  const float* b_fx    = (const float*)d_in[8];
  const float* W_Uh    = (const float*)d_in[9];
  const float* b_Uh    = (const float*)d_in[10];
  float* out = (float*)d_out;

  char* ws = (char*)d_ws;
  size_t off = 0;
  auto alloc = [&](size_t bytes) -> void* {
    void* p = ws + off;
    off = (off + bytes + 255) & ~(size_t)255;
    return p;
  };
  bf16_t* x_bf     = (bf16_t*)alloc((size_t)MROWS * KDIM * 2);   // 32 MB
  bf16_t* Wioux_bf = (bf16_t*)alloc((size_t)NIOU * KDIM * 2);    //  2 MB
  bf16_t* Wfx_bf   = (bf16_t*)alloc((size_t)NHID * KDIM * 2);    //  1 MB
  bf16_t* Wiouh_bf = (bf16_t*)alloc((size_t)NIOU * NHID * 2);    //  1 MB
  bf16_t* WUh_bf   = (bf16_t*)alloc((size_t)NHID * NHID * 2);    // .5 MB
  float*  iou_x    = (float*)alloc((size_t)MROWS * NIOU * 4);    // 64 MB
  float*  fx_x     = (float*)alloc((size_t)MROWS * NHID * 4);    // 32 MB
  float*  h_cur    = (float*)alloc((size_t)BATCH * NHID * 4);
  bf16_t* h_bf     = (bf16_t*)alloc((size_t)BATCH * NHID * 2);
  bf16_t* rh_bf    = (bf16_t*)alloc((size_t)BATCH * NHID * 2);
  float*  z_buf    = (float*)alloc((size_t)BATCH * NHID * 4);

  // Phase 0: pack operands to bf16 (float4-vectorized)
  k_build_x<<<(MROWS * KDIM) / (256 * 4), 256, 0, stream>>>(inputs, sememe,
                                                            x_bf);
  k_to_bf16<<<(NIOU * KDIM) / (256 * 4), 256, 0, stream>>>(W_ioux, Wioux_bf,
                                                           NIOU * KDIM);
  k_to_bf16<<<(NHID * KDIM) / (256 * 4), 256, 0, stream>>>(W_fx, Wfx_bf,
                                                           NHID * KDIM);
  k_to_bf16<<<(NIOU * NHID) / (256 * 4), 256, 0, stream>>>(W_iouh, Wiouh_bf,
                                                           NIOU * NHID);
  k_to_bf16<<<(NHID * NHID) / (256 * 4), 256, 0, stream>>>(W_Uh, WUh_bf,
                                                           NHID * NHID);

  // Phase 1: hoisted input projections over all timesteps
  dim3 g_iou(NIOU / 128, MROWS / 128);   // (8, 128)
  k_gemm_xw<<<g_iou, 256, 0, stream>>>(x_bf, Wioux_bf, b_ioux, iou_x, NIOU);
  dim3 g_fx(NHID / 128, MROWS / 128);    // (4, 128)
  k_gemm_xw<<<g_fx, 256, 0, stream>>>(x_bf, Wfx_bf, b_fx, fx_x, NHID);

  // Phase 2: sequential scan (recurrent weights L2-resident)
  k_init_h<<<(BATCH * NHID) / 256, 256, 0, stream>>>(hx, h_cur, h_bf);
  for (int t = 0; t < T_STEPS; ++t) {
    k_step_iou<<<16, 128, 0, stream>>>(
        h_bf, Wiouh_bf, b_iouh, iou_x + (size_t)t * BATCH * NIOU, h_cur,
        z_buf, rh_bf);
    k_step_h<<<8, 128, 0, stream>>>(
        rh_bf, WUh_bf, b_Uh, fx_x + (size_t)t * BATCH * NHID, z_buf, h_cur,
        h_bf, out + (size_t)t * BATCH * NHID);
  }

  // h_last tail of d_out
  (void)hipMemcpyAsync(out + (size_t)T_STEPS * BATCH * NHID, h_cur,
                       (size_t)BATCH * NHID * sizeof(float),
                       hipMemcpyDeviceToDevice, stream);
}
